// ScaledDotProductAttention_31353261261151
// MI455X (gfx1250) — compile-verified, matrix operand order checked by reading, not run on hardware
//
#include <hip/hip_runtime.h>
#include <hip/hip_bf16.h>
#include <math.h>

// Problem constants (from reference)
#define BATCH 2
#define HEADS 16
#define SEQ   2048
#define DKDIM 128
#define SCALE 0.08838834764831845f  // 1/sqrt(128)

typedef _Float16 v16h __attribute__((ext_vector_type(16)));
typedef _Float16 v8h  __attribute__((ext_vector_type(8)));
typedef float    v8f  __attribute__((ext_vector_type(8)));
typedef float    v4f  __attribute__((ext_vector_type(4)));

static constexpr size_t ELEMS = (size_t)BATCH * HEADS * SEQ * DKDIM;  // per-tensor

// ---------------------------------------------------------------------------
// Pre-pass A: f32 -> f16 conversion (vectorized, 8 elements/thread)
// ---------------------------------------------------------------------------
__global__ void cvt_f32_to_f16_kernel(const float* __restrict__ src,
                                      _Float16* __restrict__ dst, int n8) {
    int i = blockIdx.x * blockDim.x + threadIdx.x;
    if (i < n8) {
        v4f a = ((const v4f*)src)[2 * i];
        v4f b = ((const v4f*)src)[2 * i + 1];
        v8h o;
#pragma unroll
        for (int e = 0; e < 4; ++e) { o[e] = (_Float16)a[e]; o[4 + e] = (_Float16)b[e]; }
        ((v8h*)dst)[i] = o;
    }
}

// ---------------------------------------------------------------------------
// Pre-pass B: V [bh][SEQ][DK] f32 -> Vt [bh][DK][SEQ] f16, LDS-tiled transpose.
// Makes the PV B-fragment a contiguous f16 load (global_load_b128) instead of
// 16 strided d16 loads per fragment.
// ---------------------------------------------------------------------------
__global__ void transpose_v_f16_kernel(const float* __restrict__ V,
                                       _Float16* __restrict__ Vt) {
    __shared__ _Float16 tile[32][33];     // +1 pad kills LDS bank conflicts
    const int bh = blockIdx.z;
    const float* Vb  = V  + (size_t)bh * SEQ * DKDIM;
    _Float16*    Vtb = Vt + (size_t)bh * SEQ * DKDIM;
    const int k0 = blockIdx.x * 32;       // SEQ index base
    const int n0 = blockIdx.y * 32;       // DK index base
    const int x = threadIdx.x, y = threadIdx.y;
#pragma unroll
    for (int yy = y; yy < 32; yy += 8)    // read coalesced along DK
        tile[yy][x] = (_Float16)Vb[(size_t)(k0 + yy) * DKDIM + n0 + x];
    __syncthreads();
#pragma unroll
    for (int yy = y; yy < 32; yy += 8)    // write coalesced along SEQ
        Vtb[(size_t)(n0 + yy) * SEQ + k0 + x] = tile[x][yy];
}

// ---------------------------------------------------------------------------
// Fragment load: 16x32 f16 fragment gathered from row-major [idx][k] source.
// Layout per ISA 7.12.2 (16-bit A/B 16x32): lane<16 -> Kbase=0, lane>=16 -> Kbase=8;
// halves 0..7 -> K = Kbase+h ; halves 8..15 -> K = Kbase+16+(h-8).
// For T=_Float16 this is two contiguous 16B chunks (2x global_load_b128).
// ---------------------------------------------------------------------------
template <typename T>
__device__ __forceinline__ v16h ld_frag_rows(const T* __restrict__ base, int ld, int lane) {
    int idx = lane & 15;
    int kb  = (lane >> 4) << 3;
    const T* r = base + (size_t)idx * ld + kb;
    v16h f;
#pragma unroll
    for (int h = 0; h < 8; ++h) f[h]     = (_Float16)r[h];
#pragma unroll
    for (int h = 0; h < 8; ++h) f[h + 8] = (_Float16)r[16 + h];
    return f;
}

// ---------------------------------------------------------------------------
// Main SDPA kernel.
//   T=_Float16, VT=true : fast path; V argument points to transposed f16 V.
//   T=float,    VT=false: fallback; V row-major f32, converted on the fly.
// One wave owns a 16-row query tile; full 16x2048 prob row-block lives in LDS
// as f16 (64KB/wave, 4 waves/WGP -> one wave per SIMD32).
// ---------------------------------------------------------------------------
template <typename T, bool VT>
__global__ void sdpa_wmma_kernel(const T* __restrict__ Q,
                                 const T* __restrict__ K,
                                 const T* __restrict__ V,
                                 float* __restrict__ Out,
                                 float* __restrict__ Attn) {
    const int lane = threadIdx.x & 31;
    const int wave = threadIdx.x >> 5;
    const int bh   = blockIdx.y;                  // b*H + h
    const int q0   = blockIdx.x * 64 + wave * 16; // 16 query rows per wave

    const T* Qb = Q + (size_t)bh * SEQ * DKDIM;
    const T* Kb = K + (size_t)bh * SEQ * DKDIM;
    const T* Vb = V + (size_t)bh * SEQ * DKDIM;   // VT: [DK][SEQ], else [SEQ][DK]
    float* Ob = Out  + (size_t)bh * SEQ * DKDIM;
    float* Ab = Attn + (size_t)bh * SEQ * SEQ;

    extern __shared__ char smem[];
    _Float16* sP   = (_Float16*)smem + (size_t)wave * 16 * SEQ;   // 16x2048 f16
    float*    sInv = (float*)(smem + 4 * (16 * SEQ * sizeof(_Float16))) + wave * 16;

    // -------- Phase 1: S = scale * Q K^T into LDS f16 ------------------------
    v16h aq[4];
#pragma unroll
    for (int kc = 0; kc < 4; ++kc)
        aq[kc] = ld_frag_rows(Qb + (size_t)q0 * DKDIM + 32 * kc, DKDIM, lane);

    const int cn    = lane & 15;          // column-in-tile for C layout
    const int rbase = (lane >> 4) << 3;   // +8 row offset for hi half-wave

    for (int j = 0; j < SEQ / 16; ++j) {
        v8f s = {};
#pragma unroll
        for (int kc = 0; kc < 4; ++kc) {
            v16h bk = ld_frag_rows(Kb + (size_t)(16 * j) * DKDIM + 32 * kc, DKDIM, lane);
            s = __builtin_amdgcn_wmma_f32_16x16x32_f16(false, aq[kc], false, bk,
                                                       (short)0, s, false, false);
        }
#pragma unroll
        for (int i = 0; i < 8; ++i) {
            int row = i + rbase;
            sP[(size_t)row * SEQ + 16 * j + cn] = (_Float16)(s[i] * SCALE);
        }
    }

    // -------- Phase 2: softmax over each row (2 lanes per row) ---------------
    {
        const int r  = lane >> 1;
        const int c0 = (lane & 1) * (SEQ / 2);
        _Float16* rp = sP + (size_t)r * SEQ + c0;
        const v8h* rv = (const v8h*)rp;

        float mx = -3.0e38f;
        for (int t = 0; t < (SEQ / 2) / 8; ++t) {
            v8h x = rv[t];
#pragma unroll
            for (int e = 0; e < 8; ++e) mx = fmaxf(mx, (float)x[e]);
        }
        mx = fmaxf(mx, __shfl_xor(mx, 1, 32));

        float sum = 0.0f;
        v8h* rvw = (v8h*)rp;
        for (int t = 0; t < (SEQ / 2) / 8; ++t) {
            v8h x = rvw[t];
            v8h y;
#pragma unroll
            for (int e = 0; e < 8; ++e) {
                float ve = __expf((float)x[e] - mx);
                sum += ve;
                y[e] = (_Float16)ve;
            }
            rvw[t] = y;
        }
        sum += __shfl_xor(sum, 1, 32);
        if ((lane & 1) == 0) sInv[r] = 1.0f / sum;
    }

    // -------- Phase 3: stream attention probs to global (non-temporal) -------
    for (int r = 0; r < 16; ++r) {
        const float inv = sInv[r];
        float* dst = Ab + (size_t)(q0 + r) * SEQ;
        const _Float16* src = sP + (size_t)r * SEQ;
        for (int c = lane * 4; c < SEQ; c += 32 * 4) {
            v4f v;
#pragma unroll
            for (int e = 0; e < 4; ++e) v[e] = (float)src[c + e] * inv;
            __builtin_nontemporal_store(v, (v4f*)(dst + c));
        }
    }

    // -------- Phase 4: O = P_unnormalized * V, fold 1/sum at the end ---------
    v8f oacc[8];
#pragma unroll
    for (int nt = 0; nt < 8; ++nt) { v8f z = {}; oacc[nt] = z; }

    const int m  = lane & 15;
    const int kb = (lane >> 4) << 3;

    for (int kc = 0; kc < SEQ; kc += 32) {
        v16h a;
        {
            const _Float16* r = sP + (size_t)m * SEQ + kc + kb;
#pragma unroll
            for (int h = 0; h < 8; ++h) a[h]     = r[h];
#pragma unroll
            for (int h = 0; h < 8; ++h) a[h + 8] = r[16 + h];
        }
#pragma unroll
        for (int nt = 0; nt < 8; ++nt) {
            v16h b;
            if constexpr (VT) {
                // Vt row-major [DK][SEQ]: row = nt*16 + n, contiguous in k.
                b = ld_frag_rows(Vb + (size_t)(nt * 16) * SEQ + kc, SEQ, lane);
            } else {
                // B[k][n] = V[kc+K][nt*16 + n]; column stride DKDIM
                const T* c = Vb + (size_t)(kc + kb) * DKDIM + nt * 16 + m;
#pragma unroll
                for (int h = 0; h < 8; ++h) b[h]     = (_Float16)c[(size_t)h * DKDIM];
#pragma unroll
                for (int h = 0; h < 8; ++h) b[h + 8] = (_Float16)c[(size_t)(16 + h) * DKDIM];
            }
            oacc[nt] = __builtin_amdgcn_wmma_f32_16x16x32_f16(false, a, false, b,
                                                              (short)0, oacc[nt], false, false);
        }
    }

    float invr[8];
#pragma unroll
    for (int i = 0; i < 8; ++i) invr[i] = sInv[i + rbase];
#pragma unroll
    for (int nt = 0; nt < 8; ++nt) {
#pragma unroll
        for (int i = 0; i < 8; ++i) {
            int row = i + rbase;
            Ob[(size_t)(q0 + row) * DKDIM + nt * 16 + cn] = oacc[nt][i] * invr[i];
        }
    }
}

extern "C" void kernel_launch(void* const* d_in, const int* in_sizes, int n_in,
                              void* d_out, int out_size, void* d_ws, size_t ws_size,
                              hipStream_t stream) {
    (void)in_sizes; (void)n_in; (void)out_size;
    const float* Q = (const float*)d_in[0];
    const float* K = (const float*)d_in[1];
    const float* V = (const float*)d_in[2];
    float* out  = (float*)d_out;                                 // [B,H,S,DK]
    float* attn = out + ELEMS;                                   // [B,H,S,S]

    dim3 grid(SEQ / 64, BATCH * HEADS);   // 32 x 32 workgroups
    dim3 block(128);                      // 4 waves, each owns a 16-row q tile
    size_t shmem = 4 * (16 * SEQ * sizeof(_Float16))  // 4 waves * 64KB prob buffer
                 + 4 * 16 * sizeof(float);            // 4 waves * 16 invsum

    const size_t need = 3 * ELEMS * sizeof(_Float16);
    if (ws_size >= need) {
        // Fast path: Q,K -> f16; V -> transposed f16 [bh][DK][SEQ].
        _Float16* Qh = (_Float16*)d_ws;
        _Float16* Kh = Qh + ELEMS;
        _Float16* Vt = Kh + ELEMS;
        const int n8 = (int)(ELEMS / 8);
        const int cblk = 256;
        const int cgrd = (n8 + cblk - 1) / cblk;
        cvt_f32_to_f16_kernel<<<cgrd, cblk, 0, stream>>>(Q, Qh, n8);
        cvt_f32_to_f16_kernel<<<cgrd, cblk, 0, stream>>>(K, Kh, n8);
        transpose_v_f16_kernel<<<dim3(SEQ / 32, DKDIM / 32, BATCH * HEADS),
                                 dim3(32, 8), 0, stream>>>(V, Vt);
        sdpa_wmma_kernel<_Float16, true><<<grid, block, shmem, stream>>>(Qh, Kh, Vt, out, attn);
    } else {
        // Fallback: convert operands on the fly inside the kernel.
        sdpa_wmma_kernel<float, false><<<grid, block, shmem, stream>>>(Q, K, V, out, attn);
    }
}
